// MAB_MaskedAttention_35983236006149
// MI455X (gfx1250) — matrix operand order repaired and checked by reading, not compiled, not run on hardware
//
#include <hip/hip_runtime.h>
#include <hip/hip_bf16.h>
#include <stdint.h>

// ---------------- problem constants ----------------
#define B_    32
#define NQ_   512
#define NK_   512
#define DIMV  1024
#define H_    16
#define DH_   64

// ---------------- WMMA types (CDNA5 / gfx1250, wave32) ----------------
typedef __bf16 bf16_t;
typedef bf16_t v16bf __attribute__((ext_vector_type(16)));
typedef float  v8f   __attribute__((ext_vector_type(8)));

// TDM descriptor groups (this toolchain: 6-arg tensor_load_to_lds,
// signature (u32x4, i32x8, i32x4, i32x4, i32x8, i32 cpol))
typedef uint32_t u32x4 __attribute__((ext_vector_type(4)));
typedef int32_t  i32x8 __attribute__((ext_vector_type(8)));
typedef int32_t  i32x4 __attribute__((ext_vector_type(4)));

union Frag16 { uint32_t u[8]; v16bf v; };
union Acc8   { float    f[8]; v8f   v; };

static __device__ __forceinline__ uint16_t f32_to_bf16_bits(float x) {
  union { float f; uint32_t u; } a; a.f = x;
  uint32_t r = a.u + 0x7FFFu + ((a.u >> 16) & 1u);   // round-to-nearest-even
  return (uint16_t)(r >> 16);
}
static __device__ __forceinline__ uint32_t pack_bf16(float lo, float hi) {
  return (uint32_t)f32_to_bf16_bits(lo) | ((uint32_t)f32_to_bf16_bits(hi) << 16);
}

static __device__ __forceinline__ v8f wmma_bf16(const Frag16& a, const Frag16& b, v8f c) {
  return __builtin_amdgcn_wmma_f32_16x16x32_bf16(false, a.v, false, b.v, (short)0, c,
                                                 false, false);
}

// ----------------------------------------------------------------
// TDM 2-D tile load (data_size = 2 bytes) global -> LDS.
// D# packing per CDNA5 ISA §8.3/§8.4:
//  g0: [1:0]=count=1, [63:32]=lds_addr, [120:64]=global_addr, [127:126]=type=2
//  g1: [17:16]=data_size=1(2B), [79:48]=tensor_dim0, [111:80]=tensor_dim1,
//      [127:112]=tile_dim0, [143:128]=tile_dim1, [207:160]=tensor_dim0_stride
//  groups 2/3 (and trailing extended group) zero: <=2D tensor.
// ----------------------------------------------------------------
static __device__ __forceinline__ void tdm_load_2d_bf16(
    uint32_t lds_byte_addr, const void* gptr,
    uint32_t tile_d0, uint32_t tile_d1,
    uint32_t tensor_d0, uint32_t tensor_d1,
    uint64_t stride0_elems)
{
  const uint64_t ga = (uint64_t)(uintptr_t)gptr;
  u32x4 g0;
  g0.x = 0x1u;                                            // count = 1
  g0.y = lds_byte_addr;                                   // LDS dest (bytes)
  g0.z = (uint32_t)ga;                                    // global_addr[31:0]
  g0.w = (uint32_t)((ga >> 32) & 0x1FFFFFFu) | (2u << 30);// addr[56:32] | type=2
  i32x8 g1;
  g1[0] = (int32_t)(1u << 16);                            // data_size = 2 bytes
  g1[1] = (int32_t)((tensor_d0 & 0xFFFFu) << 16);         // tensor_dim0[15:0]
  g1[2] = (int32_t)((tensor_d0 >> 16) |
                    ((tensor_d1 & 0xFFFFu) << 16));       // dim0[31:16] | dim1[15:0]
  g1[3] = (int32_t)((tensor_d1 >> 16) |
                    ((tile_d0 & 0xFFFFu) << 16));         // dim1[31:16] | tile_dim0
  g1[4] = (int32_t)(tile_d1 & 0xFFFFu);                   // tile_dim1 (tile_dim2 = 0)
  g1[5] = (int32_t)(uint32_t)(stride0_elems & 0xFFFFFFFFull);
  g1[6] = (int32_t)(uint32_t)((stride0_elems >> 32) & 0xFFFFull);
  g1[7] = 0;
  const i32x4 z4 = {0, 0, 0, 0};
  const i32x8 z8 = {0, 0, 0, 0, 0, 0, 0, 0};
  __builtin_amdgcn_tensor_load_to_lds(g0, g1, z4, z4, z8, 0);
}

// ================================================================
// GEMM: C[m,n] = sum_k A[m,k] * W[n,k]   (i.e. X @ W^T)
// block tile 64x64, K-step 64, 256 threads = 8 wave32.
// ================================================================
enum { OUT_F32 = 0, OUT_BF16 = 1, OUT_BF16_T = 2, OUT_RELU_RES = 3 };

template <int MODE, bool A_IS_BF16>
__global__ __launch_bounds__(256) void gemm_xwt_kernel(
    const void*  __restrict__ Aptr,   // [M,K] f32 (or bf16 bits if A_IS_BF16)
    const float* __restrict__ W,      // [N,K] f32  (torch Linear weight)
    void*        __restrict__ Out,
    const float* __restrict__ bias,   // MODE==OUT_RELU_RES
    const float* __restrict__ resid,  // MODE==OUT_RELU_RES
    int M, int N, int K)
{
  __shared__ uint16_t Xs[64][72];     // bf16 bits, padded stride
  __shared__ uint16_t Ws[64][72];

  const int tid    = threadIdx.x;
  const int wave   = tid >> 5;
  const int lane   = tid & 31;
  const int laneN  = lane & 15;
  const int laneHi = lane >> 4;

  const int mblk  = blockIdx.x * 64;
  const int nblk  = blockIdx.y * 64;
  const int mwave = (wave & 3) * 16;
  const int nwave = (wave >> 2) * 32;

  Acc8 acc[2];
#pragma unroll
  for (int nt = 0; nt < 2; ++nt)
#pragma unroll
    for (int r = 0; r < 8; ++r) acc[nt].f[r] = 0.0f;

  for (int k0 = 0; k0 < K; k0 += 64) {
    __syncthreads();
    // ---- stage A tile (64x64) as bf16 into LDS ----
    if (A_IS_BF16) {
      const uint16_t* A16 = (const uint16_t*)Aptr;
#pragma unroll
      for (int i = 0; i < 8; ++i) {
        int e = i * 256 + tid;                 // dword id, 2048 dwords
        int row = e >> 5, cp = (e & 31) * 2;
        uint32_t d = *(const uint32_t*)&A16[(size_t)(mblk + row) * K + k0 + cp];
        *(uint32_t*)&Xs[row][cp] = d;
      }
    } else {
      const float* A32 = (const float*)Aptr;
#pragma unroll
      for (int i = 0; i < 8; ++i) {
        int e = i * 256 + tid;                 // float-pair id
        int row = e >> 5, cp = (e & 31) * 2;
        const float* p = &A32[(size_t)(mblk + row) * K + k0 + cp];
        *(uint32_t*)&Xs[row][cp] = pack_bf16(p[0], p[1]);
      }
    }
    // ---- stage W tile (64 rows of W, 64 k) ----
#pragma unroll
    for (int i = 0; i < 8; ++i) {
      int e = i * 256 + tid;
      int row = e >> 5, cp = (e & 31) * 2;
      const float* p = &W[(size_t)(nblk + row) * K + k0 + cp];
      *(uint32_t*)&Ws[row][cp] = pack_bf16(p[0], p[1]);
    }
    // prefetch next K slab (global_prefetch_b8)
    if (k0 + 64 < K) {
      if (!A_IS_BF16)
        __builtin_prefetch(&((const float*)Aptr)[(size_t)(mblk + (tid >> 2)) * K + k0 + 64], 0, 1);
      __builtin_prefetch(&W[(size_t)(nblk + (tid >> 2)) * K + k0 + 64], 0, 1);
    }
    __syncthreads();

    // ---- two 16x16x32 WMMA K-steps ----
#pragma unroll
    for (int ks = 0; ks < 2; ++ks) {
      const int kb = ks * 32;
      Frag16 af;                                   // A: 16x32 bf16
      const int arow = mwave + laneN;
#pragma unroll
      for (int j = 0; j < 8; ++j) {
        int kk = ((j < 4) ? 2 * j : 16 + 2 * (j - 4)) + (laneHi ? 8 : 0);
        af.u[j] = *(const uint32_t*)&Xs[arow][kb + kk];
      }
#pragma unroll
      for (int nt = 0; nt < 2; ++nt) {
        Frag16 bf;                                 // B: 32x16 bf16 (= W^T tile)
        const int brow = nwave + nt * 16 + laneN;
#pragma unroll
        for (int j = 0; j < 8; ++j) {
          int kk = ((j < 4) ? 2 * j : 16 + 2 * (j - 4)) + (laneHi ? 8 : 0);
          bf.u[j] = *(const uint32_t*)&Ws[brow][kb + kk];
        }
        acc[nt].v = wmma_bf16(af, bf, acc[nt].v);
      }
    }
  }

  // ---- epilogue (C layout: VGPR r -> row r + 8*laneHi, col laneN) ----
#pragma unroll
  for (int nt = 0; nt < 2; ++nt)
#pragma unroll
    for (int r = 0; r < 8; ++r) {
      const int m = mblk + mwave + r + 8 * laneHi;
      const int n = nblk + nwave + nt * 16 + laneN;
      const float c = acc[nt].f[r];
      if (MODE == OUT_F32) {
        ((float*)Out)[(size_t)m * N + n] = c;
      } else if (MODE == OUT_BF16) {
        ((uint16_t*)Out)[(size_t)m * N + n] = f32_to_bf16_bits(c);
      } else if (MODE == OUT_BF16_T) {           // V: store [b, n, k] transposed
        const int b = m / NK_, kk = m % NK_;
        ((uint16_t*)Out)[((size_t)b * DIMV + n) * NK_ + kk] = f32_to_bf16_bits(c);
      } else {                                   // relu(c + bias) + resid
        float y = c + bias[n];
        y = y > 0.0f ? y : 0.0f;
        ((float*)Out)[(size_t)m * N + n] = resid[(size_t)m * N + n] + y;
      }
    }
}

// ================================================================
// Flash-style masked attention with TDM-staged K/V tiles.
// grid = (NQ/64, H, B); 128 threads = 4 waves; wave w owns q rows
// [qblk*64 + w*16, +16).  Wave 0 drives the Tensor Data Mover to
// stage the shared 32x64 K tile and 64x32 V^T tile into LDS each
// key chunk; all waves consume them (4x global traffic reduction).
// ================================================================
__global__ __launch_bounds__(128) void attn_kernel(
    const float*    __restrict__ qf,   // [B,NQ,DIMV] f32 (projected q, residual src)
    const uint16_t* __restrict__ kbm,  // [B,NK,DIMV] bf16 bits
    const uint16_t* __restrict__ vT,   // [B,DIMV,NK] bf16 bits (transposed v)
    const int*      __restrict__ pmask,// [B,NK]
    float*          __restrict__ o1)   // [B,NQ,DIMV]
{
  __shared__ uint16_t Ks[32][64];                // key chunk:  32 keys x 64 d
  __shared__ uint16_t Vs[64][32];                // v^T chunk:  64 d x 32 keys
  __shared__ uint16_t Pbuf[4][16][34];           // per-wave P transpose scratch

  const int wave   = threadIdx.x >> 5;
  const int lane   = threadIdx.x & 31;
  const int laneN  = lane & 15;
  const int laneHi = lane >> 4;

  const int qblk = blockIdx.x, h = blockIdx.y, b = blockIdx.z;
  const int qbase = qblk * 64 + wave * 16;

  const float* qrow = qf + ((size_t)b * NQ_ + qbase) * DIMV + h * DH_;

  // A fragments for Q (d = 0..31 and 32..63), convert f32 -> bf16 on the fly
  Frag16 aq[2];
#pragma unroll
  for (int f = 0; f < 2; ++f)
#pragma unroll
    for (int j = 0; j < 8; ++j) {
      int kk = ((j < 4) ? 2 * j : 16 + 2 * (j - 4)) + (laneHi ? 8 : 0) + f * 32;
      const float* p = &qrow[(size_t)laneN * DIMV + kk];
      aq[f].u[j] = pack_bf16(p[0], p[1]);
    }

  Acc8 accd[4];
  float rm[8], rs[8];
#pragma unroll
  for (int r = 0; r < 8; ++r) { rm[r] = -INFINITY; rs[r] = 0.0f; }
#pragma unroll
  for (int nt = 0; nt < 4; ++nt)
#pragma unroll
    for (int r = 0; r < 8; ++r) accd[nt].f[r] = 0.0f;

  const uint16_t* kbase = kbm + (size_t)b * NK_ * DIMV + h * DH_;
  const uint16_t* vbase = vT + (size_t)b * DIMV * NK_ + (size_t)h * DH_ * NK_;
  const int* mrow = pmask + (size_t)b * NK_;
  const float scl = 1.0f / 32.0f;                // 1/sqrt(DIM_V=1024)

  const uint32_t ldsK = (uint32_t)(uintptr_t)&Ks[0][0];
  const uint32_t ldsV = (uint32_t)(uintptr_t)&Vs[0][0];

  for (int kt = 0; kt < NK_; kt += 32) {
    __syncthreads();                   // tiles from previous chunk fully consumed
    if (wave == 0) {
      // K tile: rows = 32 keys (stride DIMV), cols = 64 d contiguous
      tdm_load_2d_bf16(ldsK, kbase + (size_t)kt * DIMV,
                       /*tile*/ 64, 32, /*tensor*/ 64, 32, /*stride*/ DIMV);
      // V tile: rows = 64 d (stride NK), cols = 32 keys contiguous
      tdm_load_2d_bf16(ldsV, vbase + kt,
                       /*tile*/ 32, 64, /*tensor*/ 32, 64, /*stride*/ NK_);
      __builtin_amdgcn_s_wait_tensorcnt(0);
    }
    __syncthreads();                   // tiles visible to all waves

    // ---- scores: s[hf] covers key cols kt+hf*16 .. +15 ----
    Acc8 s[2];
#pragma unroll
    for (int hf = 0; hf < 2; ++hf) {
#pragma unroll
      for (int r = 0; r < 8; ++r) s[hf].f[r] = 0.0f;
      const uint16_t* krow = &Ks[hf * 16 + laneN][0];
#pragma unroll
      for (int f = 0; f < 2; ++f) {
        Frag16 bk;                               // B = K^T : (d, keycol)
#pragma unroll
        for (int j = 0; j < 8; ++j) {
          int dd = ((j < 4) ? 2 * j : 16 + 2 * (j - 4)) + (laneHi ? 8 : 0) + f * 32;
          bk.u[j] = *(const uint32_t*)&krow[dd];
        }
        s[hf].v = wmma_bf16(aq[f], bk, s[hf].v);
      }
    }

    // ---- mask (subtracted BEFORE 1/sqrt scale, per reference) ----
    float tmax[8];
#pragma unroll
    for (int r = 0; r < 8; ++r) tmax[r] = -3.0e38f;
#pragma unroll
    for (int hf = 0; hf < 2; ++hf) {
      const float mv = (float)mrow[kt + hf * 16 + laneN] * 1.0e10f;
#pragma unroll
      for (int r = 0; r < 8; ++r) {
        float sv = (s[hf].f[r] - mv) * scl;
        s[hf].f[r] = sv;
        tmax[r] = fmaxf(tmax[r], sv);
      }
    }
    // row max over the 16 lanes holding this row's columns
#pragma unroll
    for (int off = 1; off < 16; off <<= 1)
#pragma unroll
      for (int r = 0; r < 8; ++r)
        tmax[r] = fmaxf(tmax[r], __shfl_xor(tmax[r], off, 32));

    // ---- online softmax update ----
    float psum[8];
#pragma unroll
    for (int r = 0; r < 8; ++r) {
      const float nm = fmaxf(rm[r], tmax[r]);
      const float sc = __expf(rm[r] - nm);
      rm[r] = nm;
      rs[r] *= sc;
#pragma unroll
      for (int nt = 0; nt < 4; ++nt) accd[nt].f[r] *= sc;
      const float p0 = __expf(s[0].f[r] - nm);
      const float p1 = __expf(s[1].f[r] - nm);
      s[0].f[r] = p0; s[1].f[r] = p1;
      psum[r] = p0 + p1;
    }
#pragma unroll
    for (int off = 1; off < 16; off <<= 1)
#pragma unroll
      for (int r = 0; r < 8; ++r) psum[r] += __shfl_xor(psum[r], off, 32);
#pragma unroll
    for (int r = 0; r < 8; ++r) rs[r] += psum[r];

    // ---- C-layout -> A-layout transpose of P via per-wave LDS ----
#pragma unroll
    for (int r = 0; r < 8; ++r) {
      const int row = r + 8 * laneHi;
      Pbuf[wave][row][laneN]      = f32_to_bf16_bits(s[0].f[r]);
      Pbuf[wave][row][laneN + 16] = f32_to_bf16_bits(s[1].f[r]);
    }
    // same-wave DS RAW: hardware DScnt wait inserted by compiler; no barrier
    Frag16 ap;
#pragma unroll
    for (int j = 0; j < 8; ++j) {
      int kk = ((j < 4) ? 2 * j : 16 + 2 * (j - 4)) + (laneHi ? 8 : 0);
      ap.u[j] = *(const uint32_t*)&Pbuf[wave][laneN][kk];
    }

    // ---- accd += P (16x32) x V (32x16 per d-tile) ----
#pragma unroll
    for (int nt = 0; nt < 4; ++nt) {
      Frag16 bv;
      const uint16_t* vrow = &Vs[nt * 16 + laneN][0];
#pragma unroll
      for (int j = 0; j < 8; ++j) {
        int kk = ((j < 4) ? 2 * j : 16 + 2 * (j - 4)) + (laneHi ? 8 : 0);
        bv.u[j] = *(const uint32_t*)&vrow[kk];
      }
      accd[nt].v = wmma_bf16(ap, bv, accd[nt].v);
    }
  }

  // ---- epilogue: o1 = q + acc / rowsum ----
#pragma unroll
  for (int nt = 0; nt < 4; ++nt)
#pragma unroll
    for (int r = 0; r < 8; ++r) {
      const int m = qbase + r + 8 * laneHi;
      const int n = h * DH_ + nt * 16 + laneN;
      const size_t idx = ((size_t)b * NQ_ + m) * DIMV + n;
      o1[idx] = qf[idx] + accd[nt].f[r] / rs[r];
    }
}

// ================================================================
// Row LayerNorm over DIMV=1024; one block (256 thr) per row.
// ================================================================
template <bool WRITE_BF16>
__global__ __launch_bounds__(256) void ln_kernel(
    const float* __restrict__ X, const float* __restrict__ g,
    const float* __restrict__ bta,
    float* __restrict__ Yf, uint16_t* __restrict__ Yb)
{
  __shared__ float red[16];
  const int row = blockIdx.x, tid = threadIdx.x;
  const float* x = X + (size_t)row * DIMV;

  float v0[4], s = 0.0f, s2 = 0.0f;
#pragma unroll
  for (int i = 0; i < 4; ++i) {
    float t = x[tid + i * 256];
    v0[i] = t; s += t; s2 += t * t;
  }
#pragma unroll
  for (int off = 1; off < 32; off <<= 1) {
    s  += __shfl_xor(s, off, 32);
    s2 += __shfl_xor(s2, off, 32);
  }
  const int wave = tid >> 5, lane = tid & 31;
  if (lane == 0) { red[wave] = s; red[wave + 8] = s2; }
  __syncthreads();
  float ts = 0.0f, ts2 = 0.0f;
#pragma unroll
  for (int w = 0; w < 8; ++w) { ts += red[w]; ts2 += red[w + 8]; }
  const float mu   = ts * (1.0f / 1024.0f);
  const float var  = ts2 * (1.0f / 1024.0f) - mu * mu;
  const float rstd = rsqrtf(var + 1e-5f);
#pragma unroll
  for (int i = 0; i < 4; ++i) {
    const int n = tid + i * 256;
    const float y = (v0[i] - mu) * rstd * g[n] + bta[n];
    Yf[(size_t)row * DIMV + n] = y;
    if (WRITE_BF16) Yb[(size_t)row * DIMV + n] = f32_to_bf16_bits(y);
  }
}

// ================================================================
// Host-side launcher
// ================================================================
extern "C" void kernel_launch(void* const* d_in, const int* in_sizes, int n_in,
                              void* d_out, int out_size, void* d_ws, size_t ws_size,
                              hipStream_t stream) {
  (void)in_sizes; (void)n_in; (void)out_size; (void)ws_size;
  const float* Q    = (const float*)d_in[0];
  const float* Kx   = (const float*)d_in[1];
  const int*   mask = (const int*)d_in[2];
  const float* Wq   = (const float*)d_in[3];
  const float* Wk   = (const float*)d_in[4];
  const float* Wv   = (const float*)d_in[5];
  const float* Wo   = (const float*)d_in[6];
  const float* bo   = (const float*)d_in[7];
  const float* g0   = (const float*)d_in[8];
  const float* b0   = (const float*)d_in[9];
  const float* g1   = (const float*)d_in[10];
  const float* b1   = (const float*)d_in[11];
  float* out = (float*)d_out;

  const int M = B_ * NQ_;          // 16384 (also B_*NK_)
  const int N = DIMV, K = DIMV;

  // workspace layout (192 MB peak, buffers reused across phases)
  char* ws = (char*)d_ws;
  float*    q_f32 = (float*)(ws);                       //  64 MB [B,NQ,DIMV]
  uint16_t* k_bf  = (uint16_t*)(ws + (64ull << 20));    //  32 MB [B,NK,DIMV]
  uint16_t* vT_bf = (uint16_t*)(ws + (96ull << 20));    //  32 MB [B,DIMV,NK]
  float*    o1    = (float*)(ws + (128ull << 20));      //  64 MB attn out
  float*    ln0_f = q_f32;                              // reuse after attention
  uint16_t* ln0_b = k_bf;                               // reuse after attention
  float*    o3    = o1;                                 // reuse after LN0 consumed o1

  dim3 blk(256);
  dim3 gg(M / 64, N / 64);         // 256 x 16

  // 1) projections (bf16 WMMA, fp32 accumulate)
  gemm_xwt_kernel<OUT_F32,     false><<<gg, blk, 0, stream>>>(Q,  Wq, q_f32, nullptr, nullptr, M, N, K);
  gemm_xwt_kernel<OUT_BF16,    false><<<gg, blk, 0, stream>>>(Kx, Wk, k_bf,  nullptr, nullptr, M, N, K);
  gemm_xwt_kernel<OUT_BF16_T,  false><<<gg, blk, 0, stream>>>(Kx, Wv, vT_bf, nullptr, nullptr, M, N, K);

  // 2) masked flash attention + q residual (TDM-staged K/V)
  dim3 ga(NQ_ / 64, H_, B_);
  attn_kernel<<<ga, dim3(128), 0, stream>>>(q_f32, k_bf, vT_bf, mask, o1);

  // 3) LayerNorm0 (fp32 + bf16 copies)
  ln_kernel<true><<<M, 256, 0, stream>>>(o1, g0, b0, ln0_f, ln0_b);

  // 4) O + relu(O @ Wo^T + bo)
  gemm_xwt_kernel<OUT_RELU_RES, true><<<gg, blk, 0, stream>>>(ln0_b, Wo, o3, bo, ln0_f, M, N, K);

  // 5) LayerNorm1 -> final output
  ln_kernel<false><<<M, 256, 0, stream>>>(o3, g1, b1, out, nullptr);
}
